// GINModel_51668456571569
// MI455X (gfx1250) — compile-verified, matrix operand order checked by reading, not run on hardware
//
#include <hip/hip_runtime.h>
#include <hip/hip_bf16.h>

// ---------------------------------------------------------------------------
// GIN (5-layer GINConv + mean pool + 2-layer readout) for MI455X / gfx1250.
// Aggregation is L2-resident atomic-bound; node MLPs use v_wmma_f32_16x16x32_bf16.
// ---------------------------------------------------------------------------

typedef __attribute__((ext_vector_type(16))) __bf16 v16bf;
typedef __attribute__((ext_vector_type(8)))  float  v8f;

union FragB {
    uint4 q[2];   // two 16-byte halves (8 bf16 each)
    v16bf v;
};

__device__ __forceinline__ unsigned short f2bf(float x) {
    unsigned int u = __float_as_uint(x);
    u += 0x7fffu + ((u >> 16) & 1u);           // round-to-nearest-even
    return (unsigned short)(u >> 16);
}

constexpr int kD  = 128;    // hidden dim
constexpr int kH5 = 640;    // 5 * D readout dim
constexpr int kTM = 32;     // node rows per block in the MLP kernel

// ---------------------------------------------------------------------------
// Per-graph node counts (for mean pooling)
// ---------------------------------------------------------------------------
__global__ void count_kernel(const int* __restrict__ batch, float* __restrict__ counts, int N_) {
    int n = blockIdx.x * blockDim.x + threadIdx.x;
    if (n < N_) atomicAdd(&counts[batch[n]], 1.0f);
}

__global__ void inv_kernel(const float* __restrict__ counts, float* __restrict__ inv, int G_) {
    int g = blockIdx.x * blockDim.x + threadIdx.x;
    if (g < G_) inv[g] = 1.0f / fmaxf(counts[g], 1.0f);
}

// ---------------------------------------------------------------------------
// Edge aggregation: agg[dst] += h[src]. One wave per edge, lane = 4 dims.
// h/agg are 51 MB each -> L2-resident on the 192 MB L2.
// ---------------------------------------------------------------------------
__global__ __launch_bounds__(256) void agg_kernel(
    const float* __restrict__ h, const int* __restrict__ src,
    const int* __restrict__ dst, float* __restrict__ agg, int E_)
{
    int e    = blockIdx.x * 8 + (threadIdx.x >> 5);
    int lane = threadIdx.x & 31;
    if (e >= E_) return;
    int s = src[e], d = dst[e];
    if (e + 8 < E_) {
        int sn = src[e + 8];
        __builtin_prefetch(&h[(size_t)sn * kD + lane * 4], 0, 1);   // global_prefetch
    }
    const float4 v = *(const float4*)(h + (size_t)s * kD + lane * 4);
    float* o = agg + (size_t)d * kD + lane * 4;
    atomicAdd(o + 0, v.x);
    atomicAdd(o + 1, v.y);
    atomicAdd(o + 2, v.z);
    atomicAdd(o + 3, v.w);
}

// ---------------------------------------------------------------------------
// Fused GIN MLP for one layer:
//   z = relu(bn((h+agg) @ W1 + b1));  hout = relu(z @ W2 + b2)
//   pooled[batch[n], layer*D + :] += hout[n,:] * invc[batch[n]]
// 256 threads = 8 waves; 32-node tile; wave w owns output columns [16w,16w+16).
// ---------------------------------------------------------------------------
__global__ __launch_bounds__(256) void gin_mlp_kernel(
    const float* __restrict__ h, const float* __restrict__ agg,
    const float* __restrict__ W1, const float* __restrict__ b1,
    const float* __restrict__ gamma, const float* __restrict__ beta,
    const float* __restrict__ W2, const float* __restrict__ b2,
    const int* __restrict__ batch, const float* __restrict__ invc,
    float* __restrict__ hout, float* __restrict__ pooled,
    float bn_scale, int layer, int N_)
{
    __shared__ unsigned short sW[kD * kD];   // bf16 transposed weights (reused W1 then W2)
    __shared__ unsigned short sA[kTM * kD];  // bf16 input tile
    __shared__ unsigned short sZ[kTM * kD];  // bf16 intermediate tile
    __shared__ int   sG[kTM];
    __shared__ float sIW[kTM];

    const int tid   = threadIdx.x;
    const int lane  = tid & 31;
    const int wave  = tid >> 5;
    const int node0 = blockIdx.x * kTM;

    // Stage A = h + agg (bf16)
    for (int i = tid; i < kTM * kD; i += 256) {
        int r = i >> 7, c = i & (kD - 1);
        int node = node0 + r;
        float v = 0.f;
        if (node < N_) v = h[(size_t)node * kD + c] + agg[(size_t)node * kD + c];
        sA[i] = f2bf(v);
    }
    for (int i = tid; i < kTM; i += 256) {
        int node = node0 + i;
        int g = (node < N_) ? batch[node] : 0;
        sG[i]  = g;
        sIW[i] = (node < N_) ? invc[g] : 0.f;
    }
    // Stage W1^T (bf16): sW[n*kD + k] = W1[k,n]
    for (int i = tid; i < kD * kD; i += 256) {
        int k = i >> 7, n = i & (kD - 1);
        sW[n * kD + k] = f2bf(W1[(size_t)k * kD + n]);
    }
    __syncthreads();

    const int khalf = (lane >> 4) << 3;   // ISA 16-bit A layout: lanes 16-31 shift K by 8
    const int arow  = lane & 15;
    const int col   = wave * 16 + (lane & 15);
    const int rbase = (lane >> 4) << 3;   // C/D layout: lanes 16-31 hold rows M=r+8

    // ---- GEMM1: Z = relu(bn(A @ W1 + b1)) -------------------------------
    {
        const float bb = b1[col];
        const float s  = bn_scale * gamma[col];
        const float t  = beta[col];
        v8f acc0, acc1;
        for (int c = 0; c < 8; ++c) { acc0[c] = bb; acc1[c] = bb; }
        for (int kk = 0; kk < kD; kk += 32) {
            FragB bfr, a0, a1;
            const unsigned short* bp = sW + (size_t)col * kD + kk + khalf;
            bfr.q[0] = *(const uint4*)bp;        bfr.q[1] = *(const uint4*)(bp + 16);
            const unsigned short* p0 = sA + (size_t)arow * kD + kk + khalf;
            a0.q[0] = *(const uint4*)p0;         a0.q[1] = *(const uint4*)(p0 + 16);
            const unsigned short* p1 = sA + (size_t)(16 + arow) * kD + kk + khalf;
            a1.q[0] = *(const uint4*)p1;         a1.q[1] = *(const uint4*)(p1 + 16);
            acc0 = __builtin_amdgcn_wmma_f32_16x16x32_bf16(false, a0.v, false, bfr.v,
                                                           (short)0, acc0, false, false);
            acc1 = __builtin_amdgcn_wmma_f32_16x16x32_bf16(false, a1.v, false, bfr.v,
                                                           (short)0, acc1, false, false);
        }
        for (int r = 0; r < 8; ++r) {
            int m = r + rbase;
            float z0 = acc0[r] * s + t; z0 = z0 > 0.f ? z0 : 0.f;
            sZ[(size_t)m * kD + col] = f2bf(z0);
            float z1 = acc1[r] * s + t; z1 = z1 > 0.f ? z1 : 0.f;
            sZ[(size_t)(16 + m) * kD + col] = f2bf(z1);
        }
    }
    __syncthreads();
    // Stage W2^T into the same LDS buffer
    for (int i = tid; i < kD * kD; i += 256) {
        int k = i >> 7, n = i & (kD - 1);
        sW[n * kD + k] = f2bf(W2[(size_t)k * kD + n]);
    }
    __syncthreads();

    // ---- GEMM2: H = relu(Z @ W2 + b2); store + weighted pooled atomics ---
    {
        const float bb = b2[col];
        v8f acc0, acc1;
        for (int c = 0; c < 8; ++c) { acc0[c] = bb; acc1[c] = bb; }
        for (int kk = 0; kk < kD; kk += 32) {
            FragB bfr, a0, a1;
            const unsigned short* bp = sW + (size_t)col * kD + kk + khalf;
            bfr.q[0] = *(const uint4*)bp;        bfr.q[1] = *(const uint4*)(bp + 16);
            const unsigned short* p0 = sZ + (size_t)arow * kD + kk + khalf;
            a0.q[0] = *(const uint4*)p0;         a0.q[1] = *(const uint4*)(p0 + 16);
            const unsigned short* p1 = sZ + (size_t)(16 + arow) * kD + kk + khalf;
            a1.q[0] = *(const uint4*)p1;         a1.q[1] = *(const uint4*)(p1 + 16);
            acc0 = __builtin_amdgcn_wmma_f32_16x16x32_bf16(false, a0.v, false, bfr.v,
                                                           (short)0, acc0, false, false);
            acc1 = __builtin_amdgcn_wmma_f32_16x16x32_bf16(false, a1.v, false, bfr.v,
                                                           (short)0, acc1, false, false);
        }
        for (int r = 0; r < 8; ++r) {
            int m0 = r + rbase;
            {
                float v = acc0[r]; v = v > 0.f ? v : 0.f;
                int node = node0 + m0;
                if (node < N_) {
                    hout[(size_t)node * kD + col] = v;
                    atomicAdd(&pooled[(size_t)sG[m0] * kH5 + layer * kD + col], v * sIW[m0]);
                }
            }
            {
                int m = 16 + m0;
                float v = acc1[r]; v = v > 0.f ? v : 0.f;
                int node = node0 + m;
                if (node < N_) {
                    hout[(size_t)node * kD + col] = v;
                    atomicAdd(&pooled[(size_t)sG[m] * kH5 + layer * kD + col], v * sIW[m]);
                }
            }
        }
    }
}

// ---------------------------------------------------------------------------
// Transpose + convert lin1_W [640,640] f32 -> bf16 Wt[n*640 + k]
// ---------------------------------------------------------------------------
__global__ void cvt_lin1_kernel(const float* __restrict__ W, unsigned short* __restrict__ Wt) {
    int idx = blockIdx.x * blockDim.x + threadIdx.x;
    if (idx >= kH5 * kH5) return;
    int k = idx / kH5, n = idx % kH5;
    Wt[(size_t)n * kH5 + k] = f2bf(W[idx]);
}

// ---------------------------------------------------------------------------
// hc2 = relu(pooled @ lin1_W + lin1_b). Block = 16 rows x 128 cols, 8 waves.
// ---------------------------------------------------------------------------
__global__ __launch_bounds__(256) void lin1_kernel(
    const float* __restrict__ hc, const unsigned short* __restrict__ Wt,
    const float* __restrict__ bias, float* __restrict__ out)
{
    __shared__ unsigned short sA[16 * kH5];
    const int tid  = threadIdx.x;
    const int lane = tid & 31;
    const int wave = tid >> 5;
    const int row0 = blockIdx.x * 16;
    const int col0 = blockIdx.y * 128;

    for (int i = tid; i < 16 * kH5; i += 256) {
        int r = i / kH5, c = i % kH5;
        sA[i] = f2bf(hc[(size_t)(row0 + r) * kH5 + c]);
    }
    __syncthreads();

    const int col   = col0 + wave * 16 + (lane & 15);
    const int khalf = (lane >> 4) << 3;
    const int arow  = lane & 15;
    const float bb  = bias[col];
    v8f acc;
    for (int c = 0; c < 8; ++c) acc[c] = bb;

    for (int kk = 0; kk < kH5; kk += 32) {
        FragB bfr, a;
        const unsigned short* bp = Wt + (size_t)col * kH5 + kk + khalf;   // global bf16
        bfr.q[0] = *(const uint4*)bp;   bfr.q[1] = *(const uint4*)(bp + 16);
        const unsigned short* ap = sA + (size_t)arow * kH5 + kk + khalf;
        a.q[0] = *(const uint4*)ap;     a.q[1] = *(const uint4*)(ap + 16);
        acc = __builtin_amdgcn_wmma_f32_16x16x32_bf16(false, a.v, false, bfr.v,
                                                      (short)0, acc, false, false);
    }
    const int rbase = (lane >> 4) << 3;
    for (int r = 0; r < 8; ++r) {
        float v = acc[r]; v = v > 0.f ? v : 0.f;
        out[(size_t)(row0 + r + rbase) * kH5 + col] = v;
    }
}

// ---------------------------------------------------------------------------
// out[g] = hc2[g,:] . lin2_W + lin2_b   (wave-per-graph reduction)
// ---------------------------------------------------------------------------
__global__ __launch_bounds__(256) void lin2_kernel(
    const float* __restrict__ hc2, const float* __restrict__ w,
    const float* __restrict__ b, float* __restrict__ out, int G_)
{
    int g    = blockIdx.x * 8 + (threadIdx.x >> 5);
    int lane = threadIdx.x & 31;
    if (g >= G_) return;
    float sum = 0.f;
    for (int k = lane; k < kH5; k += 32)
        sum += hc2[(size_t)g * kH5 + k] * w[k];
    for (int off = 16; off > 0; off >>= 1)
        sum += __shfl_down(sum, off, 32);
    if (lane == 0) out[g] = sum + b[0];
}

// ---------------------------------------------------------------------------
extern "C" void kernel_launch(void* const* d_in, const int* in_sizes, int n_in,
                              void* d_out, int out_size, void* d_ws, size_t ws_size,
                              hipStream_t stream) {
    const float* x      = (const float*)d_in[0];
    const int*   ei     = (const int*)d_in[1];
    const int*   batch  = (const int*)d_in[2];
    const float* W1     = (const float*)d_in[3];
    const float* b1     = (const float*)d_in[4];
    const float* gamma  = (const float*)d_in[5];
    const float* beta   = (const float*)d_in[6];
    const float* W2     = (const float*)d_in[7];
    const float* b2     = (const float*)d_in[8];
    const float* lin1_W = (const float*)d_in[9];
    const float* lin1_b = (const float*)d_in[10];
    const float* lin2_W = (const float*)d_in[11];
    const float* lin2_b = (const float*)d_in[12];
    float* out = (float*)d_out;

    const int N_ = 100000, E_ = 3200000, G_ = 512;
    const size_t ndBytes = (size_t)N_ * kD * sizeof(float);     // 51.2 MB

    char* ws = (char*)d_ws;
    size_t off = 0;
    float* h0     = (float*)(ws + off); off += ndBytes;
    float* h1     = (float*)(ws + off); off += ndBytes;
    float* aggp   = (float*)(ws + off); off += ndBytes;
    float* pooled = (float*)(ws + off); off += (size_t)G_ * kH5 * sizeof(float);
    float* hc2    = (float*)(ws + off); off += (size_t)G_ * kH5 * sizeof(float);
    float* counts = (float*)(ws + off); off += 256 * sizeof(float) * 2;
    float* invc   = (float*)(ws + off); off += 256 * sizeof(float) * 2;
    unsigned short* Wt = (unsigned short*)(ws + off);           // 640*640 bf16

    const int* src = ei;
    const int* dst = ei + E_;

    hipMemsetAsync(counts, 0, G_ * sizeof(float), stream);
    hipMemsetAsync(pooled, 0, (size_t)G_ * kH5 * sizeof(float), stream);

    count_kernel<<<(N_ + 255) / 256, 256, 0, stream>>>(batch, counts, N_);
    inv_kernel<<<(G_ + 255) / 256, 256, 0, stream>>>(counts, invc, G_);

    const float bn_scale = 0.9999950000374997f;   // 1/sqrt(1 + 1e-5)

    const float* cur = x;
    float* ping = h0;
    float* pong = h1;
    for (int i = 0; i < 5; ++i) {
        hipMemsetAsync(aggp, 0, ndBytes, stream);
        agg_kernel<<<(E_ + 7) / 8, 256, 0, stream>>>(cur, src, dst, aggp, E_);
        gin_mlp_kernel<<<(N_ + kTM - 1) / kTM, 256, 0, stream>>>(
            cur, aggp,
            W1 + (size_t)i * kD * kD, b1 + (size_t)i * kD,
            gamma + (size_t)i * kD,  beta + (size_t)i * kD,
            W2 + (size_t)i * kD * kD, b2 + (size_t)i * kD,
            batch, invc, ping, pooled, bn_scale, i, N_);
        cur = ping;
        float* t = ping; ping = pong; pong = t;
    }

    cvt_lin1_kernel<<<(kH5 * kH5 + 255) / 256, 256, 0, stream>>>(lin1_W, Wt);
    dim3 g1(G_ / 16, kH5 / 128);
    lin1_kernel<<<g1, 256, 0, stream>>>(pooled, Wt, lin1_b, hc2);
    lin2_kernel<<<(G_ + 7) / 8, 256, 0, stream>>>(hc2, lin2_W, lin2_b, out, G_);
}
